// MultiHeadAttention_35588099015088
// MI455X (gfx1250) — matrix operand order repaired and checked by reading, not compiled, not run on hardware
//
#include <hip/hip_runtime.h>

#define D_MODEL 1024
#define N_HEAD  16
#define DH      64
#define BATCH   2
#define SEQ     2048
#define MROWS   (BATCH*SEQ)   // 4096 rows in all [B*S, D] matrices

typedef __attribute__((ext_vector_type(16))) _Float16 v16h;
typedef __attribute__((ext_vector_type(8)))  float    v8f;
typedef __attribute__((ext_vector_type(4)))  unsigned int u32x4;
typedef __attribute__((ext_vector_type(8)))  int      i32x8;
typedef __attribute__((ext_vector_type(4)))  int      i32x4;
typedef unsigned short u16;
typedef unsigned int   u32;

union U16H { v16h v; u32 u[8]; };

__device__ __forceinline__ u16 f2h(float f) {
  _Float16 h = (_Float16)f;
  return __builtin_bit_cast(u16, h);
}
__device__ __forceinline__ u32 pack2(float a, float b) {
  return (u32)f2h(a) | ((u32)f2h(b) << 16);
}
// A-matrix (16x32 f16) per-lane K pattern: VGPR v holds K = {k0, k0+1}
// lanes 0-15: v<4 -> 2v, v>=4 -> 16+2(v-4); lanes 16-31: +8  (ISA 7.12.2)
__device__ __forceinline__ int kpatA(int v, int lane) {
  return (v < 4 ? 2 * v : 16 + 2 * (v - 4)) + ((lane & 16) ? 8 : 0);
}
// B-matrix (32x16 f16) per-lane K pattern: same as A:
// lanes 0-15: v<4 -> 2v, v>=4 -> 16+2(v-4); lanes 16-31: +8
__device__ __forceinline__ int kpatB(int v, int lane) {
  return (v < 4 ? 2 * v : 16 + 2 * (v - 4)) + ((lane & 16) ? 8 : 0);
}

__device__ __forceinline__ v8f wmma_f16(v16h a, v16h b, v8f c) {
  return __builtin_amdgcn_wmma_f32_16x16x32_f16(false, a, false, b, (short)0, c,
                                                false, false);
}

// Generic->LDS offset: for the shared aperture, addr[31:0] is the LDS offset
// (ISA 00_overview 10.2 aperture rules); truncation yields the DS address.
__device__ __forceinline__ u32 lds_off(const void* p) {
  return (u32)(uintptr_t)p;
}
// gfx1250 async copy global -> LDS, 16B per lane; tracked by ASYNCcnt.
__device__ __forceinline__ void async_b128(u16* ldst, const u16* gsrc) {
  asm volatile("global_load_async_to_lds_b128 %0, %1, off"
               :: "v"(lds_off(ldst)), "v"(gsrc) : "memory");
}
__device__ __forceinline__ void wait_async0() {
  asm volatile("s_wait_asynccnt 0x0" ::: "memory");
}

// ---------------------------------------------------------------------------
// TDM 2D tile load: global [tileH rows x tileW halfs, row stride = `stride`
// halfs] -> LDS at lds_dst with per-row padding (pad codes per D# group1).
// D# built per ISA 08_async_tensor 8.3/8.4; groups 2/3 zero (2D tile).
// Issue from ONE wave (EXEC-independent DMA), wait TENSORcnt, then barrier.
// This toolchain exposes the 6-arg builtin (clang-23 / therock-10.0):
//   (uint32x4 g0, int32x8 g1, int32x4 g2, int32x4 g3, int32x8, i32 cpol)
// ---------------------------------------------------------------------------
__device__ __forceinline__ void tdm_load_2d(u16* lds_dst, const u16* gsrc,
                                            u32 tileW, u32 tileH, u32 stride,
                                            u32 padIntervalCode, u32 padAmountCode) {
  unsigned long long ga = (unsigned long long)(uintptr_t)gsrc;
  u32x4 g0;
  g0.x = 1u;                                // count=1, user descriptor
  g0.y = lds_off(lds_dst);                  // lds_addr
  g0.z = (u32)ga;                           // global_addr[31:0]
  g0.w = (u32)(ga >> 32) | (2u << 30);      // global_addr[56:32] | type=2
  i32x8 g1;
  g1[0] = (int)((1u << 16) |                // data_size: 2 bytes
                (1u << 20) |                // pad_enable
                (padIntervalCode << 22) |   // DWORDs between pads
                (padAmountCode << 25));     // pad DWORDs
  g1[1] = (int)((tileW & 0xFFFFu) << 16);                              // tensor_dim0 lo
  g1[2] = (int)(((tileW >> 16) & 0xFFFFu) | ((tileH & 0xFFFFu) << 16));// dim0 hi|dim1 lo
  g1[3] = (int)(((tileH >> 16) & 0xFFFFu) | ((tileW & 0xFFFFu) << 16));// dim1 hi|tile_dim0
  g1[4] = (int)(tileH & 0xFFFFu);           // tile_dim1 (tile_dim2 = 0)
  g1[5] = (int)stride;                      // tensor_dim0_stride[31:0]
  g1[6] = 0;                                // stride hi / dim1_stride lo
  g1[7] = 0;
  i32x4 zz4 = {0, 0, 0, 0};
  i32x8 zz8 = {0, 0, 0, 0, 0, 0, 0, 0};
  __builtin_amdgcn_tensor_load_to_lds(g0, g1, zz4, zz4, zz8, 0);
}

// ---------------------------------------------------------------------------
// fp32 -> fp16 elementwise conversion (weights)
// ---------------------------------------------------------------------------
__global__ __launch_bounds__(256) void cvt_f32_f16(const float* __restrict__ src,
                                                   u16* __restrict__ dst, int n) {
  int i = blockIdx.x * blockDim.x + threadIdx.x;
  int stride = gridDim.x * blockDim.x;
  for (; i < n; i += stride) dst[i] = f2h(src[i]);
}

// ---------------------------------------------------------------------------
// Tiled WMMA GEMM:  C[M,N] = A[M,K] * W[K,N] + bias
// A either fp32 (converted while staging) or fp16 (async-copied to LDS).
// 256 thr = 8 waves. Block tile 128(M) x 128(N), K step 32.
// Wave tile 32(M) x 64(N): 8 WMMAs per K step, B-frags reused by 2 A-frags.
// Requires M%128==0, N%128==0, K%32==0 (true here: 4096/1024/1024).
// ---------------------------------------------------------------------------
template <bool AF32>
__global__ __launch_bounds__(256) void gemm_ws(const void* __restrict__ Ap,
                                               const u16* __restrict__ Wh,
                                               const float* __restrict__ bias,
                                               u16* __restrict__ outH,
                                               float* __restrict__ outF,
                                               int M, int N, int K, int outIsF32) {
  __shared__ __align__(16) u16 Alds[128 * 40];   // [row][k], stride 40 (pad)
  __shared__ __align__(16) u16 WldsT[128 * 40];  // [n][k], transposed, stride 40
  const int tid  = threadIdx.x;
  const int lane = tid & 31;
  const int wave = tid >> 5;
  const int mBase = blockIdx.y * 128;
  const int nBase = blockIdx.x * 128;
  const int waveM = (wave & 3) * 32;
  const int waveN = (wave >> 2) * 64;

  const v8f zero = {};
  v8f acc[2][4];
#pragma unroll
  for (int m = 0; m < 2; m++)
#pragma unroll
    for (int t = 0; t < 4; t++) acc[m][t] = zero;

  for (int kt = 0; kt < K; kt += 32) {
    // ---- stage A tile 128x32 into LDS ----
    if (AF32) {
      const float* A = (const float*)Ap;
#pragma unroll
      for (int i = 0; i < 4; i++) {
        int lin = tid + 256 * i;
        int row = lin >> 3, q4 = lin & 7;  // 8 x float4 per row
        float4 f = *(const float4*)(A + (size_t)(mBase + row) * K + kt + q4 * 4);
        u32* dst = (u32*)&Alds[row * 40 + q4 * 4];
        dst[0] = pack2(f.x, f.y);
        dst[1] = pack2(f.z, f.w);
      }
    } else {
      const u16* A = (const u16*)Ap;
#pragma unroll
      for (int i = 0; i < 2; i++) {
        int lin = tid + 256 * i;
        int row = lin >> 2, q4 = lin & 3;  // 4 x 16B per row
        async_b128(&Alds[row * 40 + q4 * 8],
                   A + (size_t)(mBase + row) * K + kt + q4 * 8);
      }
    }
    // ---- stage W tile 32x128, transposed into LDS ----
#pragma unroll
    for (int i = 0; i < 2; i++) {
      int lin = tid + 256 * i;
      int k = lin >> 4, q4 = lin & 15;  // 16 x uint4 per k-row
      uint4 w = *(const uint4*)(Wh + (size_t)(kt + k) * N + nBase + q4 * 8);
      const u16* pw = (const u16*)&w;
#pragma unroll
      for (int e = 0; e < 8; e++) WldsT[(q4 * 8 + e) * 40 + k] = pw[e];
    }
    // prefetch next K tile of W into cache (global_prefetch_b8)
    if (kt + 32 < K) {
      __builtin_prefetch(Wh + (size_t)(kt + 32 + (tid >> 4)) * N + nBase +
                             (tid & 15) * 8, 0, 3);
    }
    if (!AF32) wait_async0();
    __syncthreads();

    U16H a[2];
#pragma unroll
    for (int m = 0; m < 2; m++)
#pragma unroll
      for (int v = 0; v < 8; v++)
        a[m].u[v] = *(const u32*)&Alds[(waveM + m * 16 + (lane & 15)) * 40 +
                                       kpatA(v, lane)];
#pragma unroll
    for (int t = 0; t < 4; t++) {
      U16H b;
      int n = waveN + t * 16 + (lane & 15);
#pragma unroll
      for (int v = 0; v < 8; v++)
        b.u[v] = *(const u32*)&WldsT[n * 40 + kpatB(v, lane)];
#pragma unroll
      for (int m = 0; m < 2; m++)
        acc[m][t] = wmma_f16(a[m].v, b.v, acc[m][t]);
    }
    __syncthreads();
  }

#pragma unroll
  for (int m = 0; m < 2; m++) {
#pragma unroll
    for (int t = 0; t < 4; t++) {
#pragma unroll
      for (int r = 0; r < 8; r++) {
        int row = mBase + waveM + m * 16 + r + 8 * (lane >> 4);
        int col = nBase + waveN + t * 16 + (lane & 15);
        float val = acc[m][t][r] + bias[col];
        if (outIsF32) outF[(size_t)row * N + col] = val;
        else          outH[(size_t)row * N + col] = f2h(val);
      }
    }
  }
}

// ---------------------------------------------------------------------------
// Flash attention. Layouts are [B,S,H,Dh] (== [B*S, D] from projections).
// Block = 128 thr (4 waves); each wave owns 16 query rows; block shares
// K/V LDS tiles of 32 keys. K tile staged via TDM (TENSORcnt, wave 0 issues
// the DMA descriptor; pad codes reproduce the 72-half LDS row stride);
// V tile needs transpose so it is staged through VALU. Online softmax.
// ---------------------------------------------------------------------------
__global__ __launch_bounds__(128) void attn_flash(const u16* __restrict__ Q,
                                                  const u16* __restrict__ Kp,
                                                  const u16* __restrict__ Vp,
                                                  u16* __restrict__ O) {
  __shared__ __align__(16) u16 Klds[32 * 72];    // [key][dh], stride 72
  __shared__ __align__(16) u16 VldsT[64 * 40];   // [dh][key], stride 40
  __shared__ __align__(16) u16 Plds[4 * 16 * 40];// per-wave 16x32 P tile
  const int tid  = threadIdx.x;
  const int lane = tid & 31;
  const int wave = tid >> 5;
  const int h = blockIdx.y;
  const int b = blockIdx.z;
  const int qRow0 = blockIdx.x * 64 + wave * 16;
  const size_t rowBase = (size_t)b * SEQ;
  const float scale = 0.125f;  // 1/sqrt(64)

  // preload Q fragments in A layout (2 chunks of K=32 over Dh=64)
  U16H aq[2];
  {
    int qs = qRow0 + (lane & 15);
    const u16* qrow = Q + (rowBase + qs) * D_MODEL + h * DH;
#pragma unroll
    for (int c = 0; c < 2; c++)
#pragma unroll
      for (int v = 0; v < 8; v++)
        aq[c].u[v] = *(const u32*)(qrow + c * 32 + kpatA(v, lane));
  }

  const v8f zero = {};
  v8f accO[4];
#pragma unroll
  for (int t = 0; t < 4; t++) accO[t] = zero;
  float mrow[8], lrow[8];
#pragma unroll
  for (int r = 0; r < 8; r++) { mrow[r] = -3.0e30f; lrow[r] = 0.0f; }

  u16* Pw = Plds + wave * 16 * 40;

  for (int kb = 0; kb < SEQ / 32; kb++) {
    const int keyBase = kb * 32;
    // ---- stage K via TDM: 32 rows x 64 halfs, row stride D_MODEL halfs;
    //      LDS pad: interval 32 DWORDs (code 4), amount 4 DWORDs (code 3)
    //      -> LDS row stride 72 halfs, matching Klds. One wave issues. ----
    if (wave == 0) {
      tdm_load_2d(Klds, Kp + (rowBase + keyBase) * D_MODEL + h * DH,
                  /*tileW=*/64, /*tileH=*/32, /*stride=*/D_MODEL,
                  /*padIntervalCode=*/4, /*padAmountCode=*/3);
    }
    // ---- stage V (VALU, transposed) ----
#pragma unroll
    for (int i = 0; i < 2; i++) {
      int lin = tid + 128 * i;
      int row = lin >> 3, q4 = lin & 7;  // 8 x 16B per 64-elem row
      const u16* srcv = Vp + (rowBase + keyBase + row) * D_MODEL + h * DH + q4 * 8;
      uint4 vv = *(const uint4*)srcv;
      const u16* pv = (const u16*)&vv;
#pragma unroll
      for (int e = 0; e < 8; e++) VldsT[(q4 * 8 + e) * 40 + row] = pv[e];
    }
    if (wave == 0) __builtin_amdgcn_s_wait_tensorcnt(0);
    __syncthreads();

    // ---- S = Q K^T : 16 q-rows x 32 keys (two 16-col tiles) ----
    v8f s0 = zero, s1 = zero;
#pragma unroll
    for (int c = 0; c < 2; c++) {
      U16H b0, b1;
#pragma unroll
      for (int v = 0; v < 8; v++) {
        int kk0 = kpatB(v, lane);
        b0.u[v] = *(const u32*)&Klds[(0 * 16 + (lane & 15)) * 72 + c * 32 + kk0];
        b1.u[v] = *(const u32*)&Klds[(1 * 16 + (lane & 15)) * 72 + c * 32 + kk0];
      }
      s0 = wmma_f16(aq[c].v, b0.v, s0);
      s1 = wmma_f16(aq[c].v, b1.v, s1);
    }

    // ---- online softmax; butterflies stay within 16-lane halves ----
    float p0[8], p1[8];
#pragma unroll
    for (int r = 0; r < 8; r++) {
      float v0 = s0[r] * scale, v1 = s1[r] * scale;
      float tm = fmaxf(v0, v1);
#pragma unroll
      for (int msk = 1; msk <= 8; msk <<= 1)
        tm = fmaxf(tm, __shfl_xor(tm, msk, 32));
      float mn = fmaxf(mrow[r], tm);
      float al = __expf(mrow[r] - mn);
      p0[r] = __expf(v0 - mn);
      p1[r] = __expf(v1 - mn);
      float rs = p0[r] + p1[r];
#pragma unroll
      for (int msk = 1; msk <= 8; msk <<= 1) rs += __shfl_xor(rs, msk, 32);
      lrow[r] = lrow[r] * al + rs;
      mrow[r] = mn;
#pragma unroll
      for (int t = 0; t < 4; t++) accO[t][r] *= al;
    }

    // ---- P: C layout -> LDS row-major -> regather in A layout ----
#pragma unroll
    for (int r = 0; r < 8; r++) {
      int prow = r + 8 * (lane >> 4);
      Pw[prow * 40 + (lane & 15)]      = f2h(p0[r]);
      Pw[prow * 40 + 16 + (lane & 15)] = f2h(p1[r]);
    }
    asm volatile("s_wait_dscnt 0x0" ::: "memory");  // same-wave DS ordering
    U16H ap;
#pragma unroll
    for (int v = 0; v < 8; v++)
      ap.u[v] = *(const u32*)&Pw[(lane & 15) * 40 + kpatA(v, lane)];

    // ---- O += P V (K = 32 keys, 4 n-tiles over Dh=64) ----
#pragma unroll
    for (int t = 0; t < 4; t++) {
      U16H bv;
      int ndh = t * 16 + (lane & 15);
#pragma unroll
      for (int v = 0; v < 8; v++)
        bv.u[v] = *(const u32*)&VldsT[ndh * 40 + kpatB(v, lane)];
      accO[t] = wmma_f16(ap.v, bv.v, accO[t]);
    }
    __syncthreads();
  }

  // ---- normalize and store to [B,S,H,Dh] fp16 ----
#pragma unroll
  for (int t = 0; t < 4; t++) {
#pragma unroll
    for (int r = 0; r < 8; r++) {
      int s = qRow0 + r + 8 * (lane >> 4);
      int col = t * 16 + (lane & 15);
      float val = accO[t][r] / lrow[r];
      O[(rowBase + s) * D_MODEL + h * DH + col] = f2h(val);
    }
  }
}

// ---------------------------------------------------------------------------
extern "C" void kernel_launch(void* const* d_in, const int* in_sizes, int n_in,
                              void* d_out, int out_size, void* d_ws, size_t ws_size,
                              hipStream_t stream) {
  (void)in_sizes; (void)n_in; (void)out_size; (void)ws_size;
  const float* q   = (const float*)d_in[0];
  const float* k   = (const float*)d_in[1];
  const float* v   = (const float*)d_in[2];
  const float* w_q = (const float*)d_in[3];
  const float* b_q = (const float*)d_in[4];
  const float* w_k = (const float*)d_in[5];
  const float* b_k = (const float*)d_in[6];
  const float* w_v = (const float*)d_in[7];
  const float* b_v = (const float*)d_in[8];
  const float* w_o = (const float*)d_in[9];
  const float* b_o = (const float*)d_in[10];
  float* out = (float*)d_out;

  const size_t WSZ = (size_t)D_MODEL * D_MODEL;  // 1M halfs
  const size_t PSZ = (size_t)MROWS * D_MODEL;    // 4M halfs
  u16* ws   = (u16*)d_ws;                        // ~42 MB total
  u16* wqh  = ws;
  u16* wkh  = wqh + WSZ;
  u16* wvh  = wkh + WSZ;
  u16* woh  = wvh + WSZ;
  u16* qpj  = woh + WSZ;
  u16* kpj  = qpj + PSZ;
  u16* vpj  = kpj + PSZ;
  u16* attn = vpj + PSZ;

  cvt_f32_f16<<<1024, 256, 0, stream>>>(w_q, wqh, (int)WSZ);
  cvt_f32_f16<<<1024, 256, 0, stream>>>(w_k, wkh, (int)WSZ);
  cvt_f32_f16<<<1024, 256, 0, stream>>>(w_v, wvh, (int)WSZ);
  cvt_f32_f16<<<1024, 256, 0, stream>>>(w_o, woh, (int)WSZ);

  dim3 gGemm(D_MODEL / 128, MROWS / 128);  // (8, 32)
  gemm_ws<true><<<gGemm, 256, 0, stream>>>(q, wqh, b_q, qpj, nullptr,
                                           MROWS, D_MODEL, D_MODEL, 0);
  gemm_ws<true><<<gGemm, 256, 0, stream>>>(k, wkh, b_k, kpj, nullptr,
                                           MROWS, D_MODEL, D_MODEL, 0);
  gemm_ws<true><<<gGemm, 256, 0, stream>>>(v, wvh, b_v, vpj, nullptr,
                                           MROWS, D_MODEL, D_MODEL, 0);

  dim3 gAttn(SEQ / 64, N_HEAD, BATCH);
  attn_flash<<<gAttn, 128, 0, stream>>>(qpj, kpj, vpj, attn);

  gemm_ws<false><<<gGemm, 256, 0, stream>>>(attn, woh, b_o, nullptr, out,
                                            MROWS, D_MODEL, D_MODEL, 1);
}
